// PyGNodeClassifier_11081015623892
// MI455X (gfx1250) — matrix operand order repaired and checked by reading, not compiled, not run on hardware
//
#include <hip/hip_runtime.h>
#include <hip/hip_bf16.h>

typedef __attribute__((ext_vector_type(16))) _Float16 v16h;
typedef __attribute__((ext_vector_type(8)))  _Float16 v8h;
typedef __attribute__((ext_vector_type(8)))  float    v8f;

// ---------------- utility kernels ----------------

__global__ void k_fill_f32(float* __restrict__ p, float v, long long n) {
    long long i = (long long)blockIdx.x * blockDim.x + threadIdx.x;
    if (i < n) p[i] = v;
}

__global__ void k_cvt_f32_f16(const float* __restrict__ s, _Float16* __restrict__ d, long long n) {
    long long i = (long long)blockIdx.x * blockDim.x + threadIdx.x;
    if (i < n) d[i] = (_Float16)s[i];
}

__device__ __forceinline__ void atomAddF(float* p, float v) {
    unsafeAtomicAdd(p, v);   // global_atomic_add_f32 on gfx1250
}

// degree accumulation over dst (deg pre-filled with 1.0 for self loops)
__global__ void k_deg(const int* __restrict__ dst, float* __restrict__ deg, int E) {
    int i = blockIdx.x * blockDim.x + threadIdx.x;
    if (i < E) atomAddF(deg + dst[i], 1.0f);
}

__global__ void k_dinv(const float* __restrict__ deg, float* __restrict__ dinv, int n) {
    int i = blockIdx.x * blockDim.x + threadIdx.x;
    if (i < n) dinv[i] = __frsqrt_rn(deg[i]);   // deg >= 1 always (self-loop)
}

// out[i] = bias[i % F]  (initialize accumulation target with bias)
__global__ void k_init_bias(float* __restrict__ o, const float* __restrict__ b, long long n, int F) {
    long long i = (long long)blockIdx.x * blockDim.x + threadIdx.x;
    if (i < n) o[i] = b[i % F];
}

// Y1h = f16(relu(O1 + b1))
__global__ void k_bias_relu_cvt(const float* __restrict__ X, const float* __restrict__ bias,
                                _Float16* __restrict__ Y, long long n, int F) {
    long long i = (long long)blockIdx.x * blockDim.x + threadIdx.x;
    if (i < n) Y[i] = (_Float16)fmaxf(X[i] + bias[i % F], 0.0f);
}

// ---------------- weight swizzle into WMMA B-fragment order -----------------
// Ws[((tn*KC + c)*32 + lane)*16 + j] = f16( W[(32c + (lane>>4)*16 + j)*N + 16*tn + (lane&15)] )
// so each lane's 16-half B fragment for (tile tn, K-chunk c) is contiguous (32B).
__global__ void k_swz_w(const float* __restrict__ W, _Float16* __restrict__ Ws, int N, int K) {
    int i = blockIdx.x * blockDim.x + threadIdx.x;
    if (i >= N * K) return;
    int j    = i & 15;
    int l    = (i >> 4) & 31;
    int rest = i >> 9;
    int KC   = K >> 5;
    int c    = rest % KC;
    int tn   = rest / KC;
    int row  = 32 * c + ((l >> 4) << 4) + j;
    int col  = (tn << 4) + (l & 15);
    Ws[i] = (_Float16)W[row * N + col];
}

// ---------------- WMMA GEMM: C[M, 16*TN] = A[M,K] * B, f16 in / f32 out ----
// One wave computes a 16 x (16*TN) output strip: A fragments (all K) loaded
// once and reused across all TN column tiles; B read pre-swizzled (b128 loads).
template <int K, int TN>
__global__ void k_gemm_f16(const _Float16* __restrict__ A,   // [M,K] row-major
                           const _Float16* __restrict__ Bs,  // swizzled [TN][K/32][32][16]
                           float* __restrict__ C,            // [M,16*TN] row-major
                           int numRowTiles) {
    constexpr int N  = 16 * TN;
    constexpr int KC = K / 32;
    const int lane = threadIdx.x & 31;
    const int wave = threadIdx.x >> 5;
    const int tm   = blockIdx.x * (blockDim.x >> 5) + wave;
    if (tm >= numRowTiles) return;          // wave-uniform: EXEC all-ones below
    const int lanelo = lane & 15;
    const int lanehi = lane >> 4;
    const int m = (tm << 4) + lanelo;       // A-fragment row held by this lane

    // A fragment 16x32 f16 per chunk: lanes 0-15 hold K=[kc..kc+15];
    // lanes 16-31 hold K=[kc+16..kc+31]
    v16h a[KC];
    const _Float16* arow = A + (size_t)m * K + lanehi * 16;
#pragma unroll
    for (int c = 0; c < KC; ++c) {
        union { v16h v; v8h h[2]; } u;
        u.h[0] = *(const v8h*)(arow + 32 * c);
        u.h[1] = *(const v8h*)(arow + 32 * c + 8);
        a[c] = u.v;
    }

#pragma unroll
    for (int tn = 0; tn < TN; ++tn) {
        v8f acc = {};
        const _Float16* bp = Bs + ((size_t)tn * KC * 32 + lane) * 16;
#pragma unroll
        for (int c = 0; c < KC; ++c) {
            v16h b = *(const v16h*)(bp + (size_t)c * 32 * 16);
            acc = __builtin_amdgcn_wmma_f32_16x16x32_f16(
                /*neg_a=*/false, a[c], /*neg_b=*/false, b,
                /*c_mod=*/(short)0, acc, /*reuse_a=*/false, /*reuse_b=*/false);
        }
        // C/D layout: VGPR r -> row tm*16 + r + lanehi*8, column tn*16 + lanelo
        float* crow = C + ((size_t)(tm << 4) + (size_t)lanehi * 8) * N + (tn << 4) + lanelo;
#pragma unroll
        for (int r = 0; r < 8; ++r) crow[(size_t)r * N] = acc[r];
    }
}

// ---------------- edge scatter-add -----------------------------------------
// msg = H[src] * dinv[src]*dinv[dst]; O[dst] += msg.  Self-loops are the
// virtual edges e in [E, E+Nn).  F/4 threads per edge, float4 per thread.
template <int F>
__global__ void k_scatter(const float* __restrict__ H, float* __restrict__ O,
                          const int* __restrict__ src, const int* __restrict__ dst,
                          const float* __restrict__ dinv, int E, int Nn) {
    constexpr int CPE = F >> 2;
    long long tid   = (long long)blockIdx.x * blockDim.x + threadIdx.x;
    long long total = (long long)(E + Nn) * CPE;
    if (tid >= total) return;
    int e = (int)(tid / CPE);
    int c = (int)(tid % CPE);
    int s, d;
    if (e < E) { s = src[e]; d = dst[e]; } else { s = d = e - E; }
    float w = dinv[s] * dinv[d];
    const float4 v = *(const float4*)(H + (size_t)s * F + c * 4);
    float* o = O + (size_t)d * F + c * 4;
    atomAddF(o + 0, v.x * w);
    atomAddF(o + 1, v.y * w);
    atomAddF(o + 2, v.z * w);
    atomAddF(o + 3, v.w * w);
}

// ---------------- launcher --------------------------------------------------

static inline unsigned blks(long long n, int t = 256) { return (unsigned)((n + t - 1) / t); }

extern "C" void kernel_launch(void* const* d_in, const int* in_sizes, int n_in,
                              void* d_out, int out_size, void* d_ws, size_t ws_size,
                              hipStream_t stream) {
    const float* x  = (const float*)d_in[0];
    const int*   ei = (const int*)d_in[1];
    const float* W1 = (const float*)d_in[2];
    const float* b1 = (const float*)d_in[3];
    const float* W2 = (const float*)d_in[4];
    const float* b2 = (const float*)d_in[5];
    float* out = (float*)d_out;

    const int F1 = 128;
    const int Nn = in_sizes[0] / F1;        // 100000
    const int E  = in_sizes[1] / 2;         // 1600000
    const int F2 = in_sizes[4] / F1;        // 16
    const int* src = ei;
    const int* dst = ei + E;

    // workspace layout (256B aligned slices)
    char* ws = (char*)d_ws;
    size_t off = 0;
    auto take = [&](size_t bytes) { char* p = ws + off; off = (off + bytes + 255) & ~(size_t)255; return p; };
    _Float16* Xh  = (_Float16*)take((size_t)Nn * F1 * 2);   // reused as Y1h for layer 2
    _Float16* W1s = (_Float16*)take((size_t)F1 * F1 * 2);   // swizzled B fragments
    _Float16* W2s = (_Float16*)take((size_t)F1 * F2 * 2);
    float*    deg = (float*)take((size_t)Nn * 4);
    float*    dnv = (float*)take((size_t)Nn * 4);
    float*    H1  = (float*)take((size_t)Nn * F1 * 4);
    float*    O1  = (float*)take((size_t)Nn * F1 * 4);
    float*    H2  = (float*)take((size_t)Nn * F2 * 4);
    (void)ws_size;

    const long long nXf = (long long)Nn * F1;
    const int rowTiles = Nn / 16;           // 6250 (exact)

    // weights -> f16 fragment-swizzled layout
    k_swz_w<<<blks((long long)F1 * F1), 256, 0, stream>>>(W1, W1s, F1, F1);
    k_swz_w<<<blks((long long)F1 * F2), 256, 0, stream>>>(W2, W2s, F2, F1);
    // features -> f16
    k_cvt_f32_f16<<<blks(nXf), 256, 0, stream>>>(x, Xh, nXf);

    // symmetric normalization weights
    k_fill_f32<<<blks(Nn), 256, 0, stream>>>(deg, 1.0f, Nn);           // self-loop
    k_deg<<<blks(E), 256, 0, stream>>>(dst, deg, E);
    k_dinv<<<blks(Nn), 256, 0, stream>>>(deg, dnv, Nn);

    // layer 1: H1 = x @ W1  (one wave per 16x128 strip, 32 wmma each)
    k_gemm_f16<128, 8><<<blks((long long)rowTiles * 32), 256, 0, stream>>>(Xh, W1s, H1, rowTiles);
    k_fill_f32<<<blks(nXf), 256, 0, stream>>>(O1, 0.0f, nXf);
    k_scatter<128><<<blks((long long)(E + Nn) * (128 / 4)), 256, 0, stream>>>(H1, O1, src, dst, dnv, E, Nn);
    k_bias_relu_cvt<<<blks(nXf), 256, 0, stream>>>(O1, b1, Xh, nXf, F1);   // Xh now holds Y1h

    // layer 2: H2 = relu(out1) @ W2  (one wave per 16x16 strip, 4 wmma each)
    k_gemm_f16<128, 1><<<blks((long long)rowTiles * 32), 256, 0, stream>>>(Xh, W2s, H2, rowTiles);
    k_init_bias<<<blks((long long)Nn * F2), 256, 0, stream>>>(out, b2, (long long)Nn * F2, F2);
    k_scatter<16><<<blks((long long)(E + Nn) * (16 / 4)), 256, 0, stream>>>(H2, out, src, dst, dnv, E, Nn);
}
